// FasterRCNN_7567732376311
// MI455X (gfx1250) — matrix operand order repaired and checked by reading, not compiled, not run on hardware
//
#include <hip/hip_runtime.h>
#include <hip/hip_bf16.h>

// ---------------------------------------------------------------------------
// Types for CDNA5 WMMA / TDM builtins
// ---------------------------------------------------------------------------
typedef __attribute__((ext_vector_type(16))) __bf16     v16bf;
typedef __attribute__((ext_vector_type(8)))  float      v8f;
typedef __attribute__((ext_vector_type(4)))  unsigned   v4u;
typedef __attribute__((ext_vector_type(8)))  int        v8i;
typedef __attribute__((ext_vector_type(4)))  int        v4i;

// Problem constants (match reference)
constexpr int BB      = 2;
constexpr int CCH     = 512;
constexpr int HF      = 62;
constexpr int WF      = 62;
constexpr int NPIX    = HF * WF;          // 3844
constexpr int NANC    = NPIX * 9;         // 34596
constexpr int NSORT   = 65536;            // pow2 pad for bitonic
constexpr int PRE_NMS = 6000;
constexpr int POST    = 300;
constexpr int RTOT    = BB * POST;        // 600
constexpr int FEATK   = CCH * 49;         // 25088
constexpr int HID     = 4096;
constexpr int NCLS    = 21;
constexpr int OUTC    = NCLS * 5;         // 105 (cls 21 + loc 84)

__device__ __forceinline__ __bf16 f2bf(float f) { return (__bf16)f; }

// 16-bit B fragment (32x16, KxN): k = half*8 + (e&7) + 16*(e>>3)  (same runs as A)
// 16-bit A fragment (16x32, MxK): k runs are [half*8 .. half*8+7] and
//                                 [16+half*8 .. 16+half*8+7]  (two float4 pairs)

// ---------------------------------------------------------------------------
// TDM: 2D fp32 tile (tile_rows x tile_cols) from row-major matrix into LDS.
// Uniform (wave-scalar) arguments only.
// ---------------------------------------------------------------------------
__device__ __forceinline__ void tdm_load_2d(unsigned lds_off, const float* gptr,
                                            unsigned long long tensor_cols,
                                            unsigned long long tensor_rows,
                                            unsigned tile_cols, unsigned tile_rows,
                                            unsigned long long row_stride_elems)
{
    unsigned long long ga = (unsigned long long)(size_t)gptr; // byte addr of tile start
    v4u g0;
    g0.x = 1u;                                     // count=1, user descriptor
    g0.y = lds_off;                                // lds_addr (bytes)
    g0.z = (unsigned)(ga & 0xFFFFFFFFull);         // global_addr[31:0]
    g0.w = (unsigned)((ga >> 32) & 0x01FFFFFFull) | (2u << 30); // addr[56:32], type=2
    v8i g1;
    g1[0] = (int)(2u << 16);                                        // data_size = 4B
    g1[1] = (int)((unsigned)(tensor_cols & 0xFFFFull) << 16);       // dim0 lo16
    g1[2] = (int)(((unsigned)((tensor_cols >> 16) & 0xFFFFull)) |
                  ((unsigned)(tensor_rows & 0xFFFFull) << 16));     // dim0 hi16 | dim1 lo16
    g1[3] = (int)(((unsigned)((tensor_rows >> 16) & 0xFFFFull)) |
                  (tile_cols << 16));                               // dim1 hi16 | tile_dim0
    g1[4] = (int)(tile_rows & 0xFFFFu);                             // tile_dim1 (tile_dim2=0)
    g1[5] = (int)(unsigned)(row_stride_elems & 0xFFFFFFFFull);      // dim0_stride lo32
    g1[6] = (int)((unsigned)((row_stride_elems >> 32) & 0xFFFFull));// dim0_stride hi16
    g1[7] = 0;
    v4i z = {0, 0, 0, 0};
#if defined(__clang_major__) && (__clang_major__ >= 23)
    v8i z8 = {0, 0, 0, 0, 0, 0, 0, 0};
    __builtin_amdgcn_tensor_load_to_lds(g0, g1, z, z, z8, 0);
#else
    __builtin_amdgcn_tensor_load_to_lds(g0, g1, z, z, 0);
#endif
}

// ---------------------------------------------------------------------------
// 0a) Weight transpose OIHW -> [tap][oc][ic]  (A fragments become contiguous)
// ---------------------------------------------------------------------------
__global__ __launch_bounds__(256) void k_wtrans(const float* __restrict__ w,
                                                float* __restrict__ wT)
{
    const int gid = blockIdx.x * 256 + threadIdx.x;   // (oc*512+ic)*9+tap order
    if (gid >= CCH * CCH * 9) return;
    const int tap = gid % 9;
    const int t   = gid / 9;
    const int ic  = t % CCH;
    const int oc  = t / CCH;
    wT[((size_t)tap * CCH + oc) * CCH + ic] = w[gid];
}

// ---------------------------------------------------------------------------
// 0b) Activation transpose NCHW -> NHWC  (B fragments become contiguous)
// ---------------------------------------------------------------------------
__global__ __launch_bounds__(256) void k_xtrans(const float* __restrict__ x,
                                                float* __restrict__ xT)
{
    const int gid = blockIdx.x * 256 + threadIdx.x;   // (b*512+c)*NPIX+p order
    if (gid >= BB * CCH * NPIX) return;
    const int p = gid % NPIX;
    const int t = gid / NPIX;
    const int c = t % CCH;
    const int b = t / CCH;
    xT[((size_t)b * NPIX + p) * CCH + c] = x[gid];
}

// ---------------------------------------------------------------------------
// 1) conv1 3x3 SAME (512->512) + ReLU, implicit GEMM via bf16 WMMA.
//    One wave per 16(OC) x 16(pixel) tile; b128 loads, maskless predication.
// ---------------------------------------------------------------------------
__global__ __launch_bounds__(32) void k_conv3x3_wmma(const float* __restrict__ xT,
                                                     const float* __restrict__ wT,
                                                     const float* __restrict__ bias,
                                                     float* __restrict__ h)
{
    const int lane = threadIdx.x & 31;
    const int half = lane >> 4, l16 = lane & 15;
    const int octile = blockIdx.x, ntile = blockIdx.y, bb = blockIdx.z;
    const int oc = octile * 16 + l16;    // A-fragment row (weights)
    const int n  = ntile * 16 + l16;     // B-fragment column (pixel)
    const bool nin = (n < NPIX);
    const int py = nin ? (n / WF) : 0;
    const int px = nin ? (n % WF) : 0;

    v8f acc = {};
    for (int tap = 0; tap < 9; ++tap) {
        const int ky = tap / 3 - 1, kx = tap % 3 - 1;
        const int iy = py + ky, ix = px + kx;
        const float bm = (nin && iy >= 0 && iy < HF && ix >= 0 && ix < WF) ? 1.0f : 0.0f;
        const int iyc = min(max(iy, 0), HF - 1);
        const int ixc = min(max(ix, 0), WF - 1);
        const float* xp = xT + ((size_t)bb * NPIX + iyc * WF + ixc) * CCH;
        const float* wp = wT + ((size_t)tap * CCH + oc) * CCH;
        for (int ic0 = 0; ic0 < CCH; ic0 += 32) {
            const float* ap = wp + ic0 + half * 8;
            const float4 a0 = *(const float4*)(ap);
            const float4 a1 = *(const float4*)(ap + 4);
            const float4 a2 = *(const float4*)(ap + 16);
            const float4 a3 = *(const float4*)(ap + 20);
            const float* bp = xp + ic0 + half * 8;
            const float4 b0 = *(const float4*)(bp);
            const float4 b1 = *(const float4*)(bp + 4);
            const float4 b2 = *(const float4*)(bp + 16);
            const float4 b3 = *(const float4*)(bp + 20);
            v16bf af, bt;
            af[0]  = f2bf(a0.x); af[1]  = f2bf(a0.y); af[2]  = f2bf(a0.z); af[3]  = f2bf(a0.w);
            af[4]  = f2bf(a1.x); af[5]  = f2bf(a1.y); af[6]  = f2bf(a1.z); af[7]  = f2bf(a1.w);
            af[8]  = f2bf(a2.x); af[9]  = f2bf(a2.y); af[10] = f2bf(a2.z); af[11] = f2bf(a2.w);
            af[12] = f2bf(a3.x); af[13] = f2bf(a3.y); af[14] = f2bf(a3.z); af[15] = f2bf(a3.w);
            bt[0]  = f2bf(b0.x * bm); bt[1]  = f2bf(b0.y * bm); bt[2]  = f2bf(b0.z * bm); bt[3]  = f2bf(b0.w * bm);
            bt[4]  = f2bf(b1.x * bm); bt[5]  = f2bf(b1.y * bm); bt[6]  = f2bf(b1.z * bm); bt[7]  = f2bf(b1.w * bm);
            bt[8]  = f2bf(b2.x * bm); bt[9]  = f2bf(b2.y * bm); bt[10] = f2bf(b2.z * bm); bt[11] = f2bf(b2.w * bm);
            bt[12] = f2bf(b3.x * bm); bt[13] = f2bf(b3.y * bm); bt[14] = f2bf(b3.z * bm); bt[15] = f2bf(b3.w * bm);
            acc = __builtin_amdgcn_wmma_f32_16x16x32_bf16(false, af, false, bt,
                                                          (short)0, acc, false, false);
        }
    }
    if (nin) {
#pragma unroll
        for (int r = 0; r < 8; ++r) {
            const int oc2 = octile * 16 + r + half * 8;
            float v = acc[r] + bias[oc2];
            h[(((size_t)bb * CCH + oc2) * HF + py) * WF + px] = fmaxf(v, 0.0f);
        }
    }
}

// ---------------------------------------------------------------------------
// 2) RPN heads (1x1 convs), softmax score, box decode, sort-key generation.
// ---------------------------------------------------------------------------
__device__ __forceinline__ unsigned ford(float f) {
    unsigned u = __float_as_uint(f);
    return (u & 0x80000000u) ? ~u : (u | 0x80000000u);
}

__global__ __launch_bounds__(256) void k_heads(const float* __restrict__ h,
                                               const float* __restrict__ anchors,
                                               const float* __restrict__ w_loc,
                                               const float* __restrict__ b_loc,
                                               const float* __restrict__ w_obj,
                                               const float* __restrict__ b_obj,
                                               float* __restrict__ boxes,
                                               unsigned long long* __restrict__ keys)
{
    const int gid = blockIdx.x * 256 + threadIdx.x;   // [0, 2*65536)
    const int b = gid >> 16;
    const int n = gid & 0xFFFF;
    unsigned long long key = 0ull;
    if (n < NANC) {
        const int p = n / 9, anc = n % 9;
        const int py = p / WF, px = p % WF;
        const float* hb = h + ((size_t)b * CCH) * NPIX + py * WF + px;
        float l0 = b_loc[anc * 4 + 0], l1 = b_loc[anc * 4 + 1];
        float l2 = b_loc[anc * 4 + 2], l3 = b_loc[anc * 4 + 3];
        float o0 = b_obj[anc * 2 + 0], o1 = b_obj[anc * 2 + 1];
        for (int c = 0; c < CCH; ++c) {
            const float hv = hb[(size_t)c * NPIX];
            l0 += hv * w_loc[(anc * 4 + 0) * CCH + c];
            l1 += hv * w_loc[(anc * 4 + 1) * CCH + c];
            l2 += hv * w_loc[(anc * 4 + 2) * CCH + c];
            l3 += hv * w_loc[(anc * 4 + 3) * CCH + c];
            o0 += hv * w_obj[(anc * 2 + 0) * CCH + c];
            o1 += hv * w_obj[(anc * 2 + 1) * CCH + c];
        }
        const float score = 1.0f / (1.0f + expf(o0 - o1));
        const float ax1 = anchors[n * 4 + 0], ay1 = anchors[n * 4 + 1];
        const float ax2 = anchors[n * 4 + 2], ay2 = anchors[n * 4 + 3];
        const float aw = ax2 - ax1, ah = ay2 - ay1;
        const float acx = ax1 + 0.5f * aw, acy = ay1 + 0.5f * ah;
        const float cx = acx + l0 * aw, cy = acy + l1 * ah;
        const float ww = aw * expf(l2), hh = ah * expf(l3);
        const float x1 = fminf(fmaxf(cx - 0.5f * ww, 0.0f), 992.0f);
        const float y1 = fminf(fmaxf(cy - 0.5f * hh, 0.0f), 992.0f);
        const float x2 = fminf(fmaxf(cx + 0.5f * ww, 0.0f), 992.0f);
        const float y2 = fminf(fmaxf(cy + 0.5f * hh, 0.0f), 992.0f);
        float* bp = boxes + ((size_t)b * NANC + n) * 4;
        bp[0] = x1; bp[1] = y1; bp[2] = x2; bp[3] = y2;
        if ((x2 - x1 >= 16.0f) && (y2 - y1 >= 16.0f))
            key = ((unsigned long long)ford(score) << 32) | (unsigned)n;
    }
    keys[((size_t)b << 16) + n] = key;
}

// ---------------------------------------------------------------------------
// 3) Grid-wide bitonic sort step (descending), 2 batches x 65536 keys
// ---------------------------------------------------------------------------
__global__ __launch_bounds__(256) void k_bitonic(unsigned long long* __restrict__ keys,
                                                 int j, int k)
{
    const int t = blockIdx.x * 256 + threadIdx.x;  // [0, 131072)
    const int b = t >> 16;
    const int i = t & 0xFFFF;
    const int ixj = i ^ j;
    if (ixj > i) {
        unsigned long long* kp = keys + ((size_t)b << 16);
        const unsigned long long a = kp[i], c = kp[ixj];
        const bool up = ((i & k) == 0);   // descending order
        if (up ? (a < c) : (a > c)) { kp[i] = c; kp[ixj] = a; }
    }
}

// ---------------------------------------------------------------------------
// 4) Greedy NMS: one workgroup per batch. Suppression bitmask in LDS.
// ---------------------------------------------------------------------------
__global__ __launch_bounds__(1024) void k_nms(const unsigned long long* __restrict__ keys,
                                              const float* __restrict__ boxes,
                                              float* __restrict__ nbx,
                                              float* __restrict__ nar,
                                              int* __restrict__ nvd,
                                              float* __restrict__ rois,
                                              float* __restrict__ rv)
{
    const int b = blockIdx.x;
    const int tid = threadIdx.x;
    const unsigned long long* kp = keys + ((size_t)b << 16);
    const float* bxs = boxes + (size_t)b * NANC * 4;
    float* tb = nbx + (size_t)b * PRE_NMS * 4;
    float* ta = nar + (size_t)b * PRE_NMS;
    int*   tv = nvd + (size_t)b * PRE_NMS;

    for (int t = tid; t < PRE_NMS; t += 1024) {
        const unsigned long long kk = kp[t];
        const int vdv = (kk >> 32) != 0ull;
        const int idx = (int)(kk & 0xFFFFFFFFull);
        float x1 = 0, y1 = 0, x2 = 0, y2 = 0;
        if (vdv) {
            x1 = bxs[idx * 4 + 0]; y1 = bxs[idx * 4 + 1];
            x2 = bxs[idx * 4 + 2]; y2 = bxs[idx * 4 + 3];
        }
        tb[t * 4 + 0] = x1; tb[t * 4 + 1] = y1;
        tb[t * 4 + 2] = x2; tb[t * 4 + 3] = y2;
        ta[t] = fmaxf(x2 - x1, 0.0f) * fmaxf(y2 - y1, 0.0f);
        tv[t] = vdv;
    }
    for (int t = tid; t < POST; t += 1024) {
        rois[((size_t)b * POST + t) * 4 + 0] = 0.0f;
        rois[((size_t)b * POST + t) * 4 + 1] = 0.0f;
        rois[((size_t)b * POST + t) * 4 + 2] = 0.0f;
        rois[((size_t)b * POST + t) * 4 + 3] = 0.0f;
        rv[(size_t)b * POST + t] = 0.0f;
    }
    __shared__ unsigned sup[(PRE_NMS + 31) / 32];
    __shared__ int kept;
    for (int t = tid; t < (PRE_NMS + 31) / 32; t += 1024) sup[t] = 0u;
    if (tid == 0) kept = 0;
    __syncthreads();

    for (int i = 0; i < PRE_NMS; ++i) {
        const bool alive = tv[i] && !((sup[i >> 5] >> (i & 31)) & 1u);
        if (alive) {
            const float xi1 = tb[i * 4 + 0], yi1 = tb[i * 4 + 1];
            const float xi2 = tb[i * 4 + 2], yi2 = tb[i * 4 + 3];
            const float ai = ta[i];
            for (int j = i + 1 + tid; j < PRE_NMS; j += 1024) {
                const float xx1 = fmaxf(xi1, tb[j * 4 + 0]);
                const float yy1 = fmaxf(yi1, tb[j * 4 + 1]);
                const float xx2 = fminf(xi2, tb[j * 4 + 2]);
                const float yy2 = fminf(yi2, tb[j * 4 + 3]);
                const float inter = fmaxf(xx2 - xx1, 0.0f) * fmaxf(yy2 - yy1, 0.0f);
                const float iou = inter / (ai + ta[j] - inter + 1e-9f);
                if (iou > 0.7f) atomicOr(&sup[j >> 5], 1u << (j & 31));
            }
            if (tid == 0 && kept < POST) {
                float* rp = rois + ((size_t)b * POST + kept) * 4;
                rp[0] = xi1; rp[1] = yi1; rp[2] = xi2; rp[3] = yi2;
                rv[(size_t)b * POST + kept] = 1.0f;
                kept = kept + 1;
            }
        }
        __syncthreads();
    }
}

// ---------------------------------------------------------------------------
// 5) ROI-align (7x7, bilinear) -> feat[600][512*49]
// ---------------------------------------------------------------------------
__global__ __launch_bounds__(256) void k_roialign(const float* __restrict__ x,
                                                  const float* __restrict__ rois,
                                                  float* __restrict__ feat)
{
    const int gid = blockIdx.x * 256 + threadIdx.x;     // 600*25088 total
    const int r = gid / FEATK;
    if (r >= RTOT) return;
    const int rem = gid - r * FEATK;
    const int c = rem / 49;
    const int pp = rem - c * 49;
    const int b = r / POST;
    const float s = 1.0f / 16.0f;
    const float x1 = rois[r * 4 + 0] * s, y1 = rois[r * 4 + 1] * s;
    const float x2 = rois[r * 4 + 2] * s, y2 = rois[r * 4 + 3] * s;
    const int iy = pp / 7, ix = pp % 7;
    float X = x1 + ((ix + 0.5f) / 7.0f) * (x2 - x1);
    float Y = y1 + ((iy + 0.5f) / 7.0f) * (y2 - y1);
    X = fminf(fmaxf(X, 0.0f), (float)(WF - 1));
    Y = fminf(fmaxf(Y, 0.0f), (float)(HF - 1));
    const float fx0 = floorf(X), fy0 = floorf(Y);
    const float lx = X - fx0, ly = Y - fy0;
    const int x0i = (int)fx0, y0i = (int)fy0;
    const int x1i = min(x0i + 1, WF - 1), y1i = min(y0i + 1, HF - 1);
    const float* xb = x + ((size_t)b * CCH + c) * NPIX;
    const float v = (1.0f - ly) * (1.0f - lx) * xb[y0i * WF + x0i]
                  + (1.0f - ly) * lx          * xb[y0i * WF + x1i]
                  + ly * (1.0f - lx)          * xb[y1i * WF + x0i]
                  + ly * lx                   * xb[y1i * WF + x1i];
    feat[(size_t)r * FEATK + c * 49 + pp] = v;
}

// ---------------------------------------------------------------------------
// 6) Blocked FC GEMM + bias + ReLU: C[M,N] = relu(A[M,K] @ B[K,N] + bias)
//    256 threads = 8 waves computing a 64x32 tile. B tile staged via TDM.
// ---------------------------------------------------------------------------
constexpr int FC_MT = 64, FC_NT = 32, FC_KT = 32;

__global__ __launch_bounds__(256) void k_fc_relu_wmma(const float* __restrict__ A, int lda,
                                                      const float* __restrict__ Bw, int ldb,
                                                      const float* __restrict__ bias,
                                                      float* __restrict__ C, int ldc,
                                                      int M, int N, int K)
{
    extern __shared__ float smem[];
    float* sB = smem;                       // FC_KT x FC_NT
    float* sA = smem + FC_KT * FC_NT;       // FC_MT x FC_KT
    const int tid = threadIdx.x;
    const int lane = tid & 31, wid = tid >> 5;
    const int wy = wid >> 1, wx = wid & 1;  // 4 x 2 waves
    const int half = lane >> 4, l16 = lane & 15;
    const int m0 = blockIdx.y * FC_MT, n0 = blockIdx.x * FC_NT;

    v8f acc = {};
    for (int k0 = 0; k0 < K; k0 += FC_KT) {
        for (int i = tid; i < FC_MT * FC_KT; i += 256) {
            const int mm = i >> 5, kk = i & 31;
            const int gm = m0 + mm;
            sA[i] = (gm < M) ? A[(size_t)gm * lda + k0 + kk] : 0.0f;
        }
        if (wid == 0) {
            tdm_load_2d(0u, Bw + (size_t)k0 * ldb + n0,
                        (unsigned long long)N, (unsigned long long)K,
                        (unsigned)FC_NT, (unsigned)FC_KT,
                        (unsigned long long)ldb);
            __builtin_amdgcn_s_wait_tensorcnt(0);
        }
        __syncthreads();
        // A fragment: two contiguous 8-float runs -> ds_load_b128 pairs
        const float* arow = sA + (wy * 16 + l16) * FC_KT + half * 8;
        const float4 q0 = *(const float4*)(arow);
        const float4 q1 = *(const float4*)(arow + 4);
        const float4 q2 = *(const float4*)(arow + 16);
        const float4 q3 = *(const float4*)(arow + 20);
        v16bf af, bt;
        af[0]  = f2bf(q0.x); af[1]  = f2bf(q0.y); af[2]  = f2bf(q0.z); af[3]  = f2bf(q0.w);
        af[4]  = f2bf(q1.x); af[5]  = f2bf(q1.y); af[6]  = f2bf(q1.z); af[7]  = f2bf(q1.w);
        af[8]  = f2bf(q2.x); af[9]  = f2bf(q2.y); af[10] = f2bf(q2.z); af[11] = f2bf(q2.w);
        af[12] = f2bf(q3.x); af[13] = f2bf(q3.y); af[14] = f2bf(q3.z); af[15] = f2bf(q3.w);
#pragma unroll
        for (int e = 0; e < 16; ++e)
            bt[e] = f2bf(sB[(half * 8 + (e & 7) + (e >> 3) * 16) * FC_NT + wx * 16 + l16]);
        acc = __builtin_amdgcn_wmma_f32_16x16x32_bf16(false, af, false, bt,
                                                      (short)0, acc, false, false);
        __syncthreads();
    }
    const int n2 = n0 + wx * 16 + l16;
#pragma unroll
    for (int r = 0; r < 8; ++r) {
        const int m2 = m0 + wy * 16 + r + half * 8;
        if (m2 < M) {
            const float v = acc[r] + bias[n2];
            C[(size_t)m2 * ldc + n2] = fmaxf(v, 0.0f);
        }
    }
}

// ---------------------------------------------------------------------------
// 7) Final small GEMMs (N=21 and N=84), concat, scale by roi-valid
// ---------------------------------------------------------------------------
__global__ __launch_bounds__(256) void k_final(const float* __restrict__ y2,
                                               const float* __restrict__ Wc,
                                               const float* __restrict__ bc,
                                               const float* __restrict__ Wl,
                                               const float* __restrict__ bl,
                                               const float* __restrict__ rv,
                                               float* __restrict__ out)
{
    const int gid = blockIdx.x * 256 + threadIdx.x;
    if (gid >= RTOT * OUTC) return;
    const int r = gid / OUTC;
    const int o = gid - r * OUTC;
    const float* yr = y2 + (size_t)r * HID;
    float acc;
    if (o < NCLS) {
        acc = bc[o];
        for (int k = 0; k < HID; ++k) acc += yr[k] * Wc[(size_t)k * NCLS + o];
    } else {
        const int oo = o - NCLS;
        acc = bl[oo];
        for (int k = 0; k < HID; ++k) acc += yr[k] * Wl[(size_t)k * (NCLS * 4) + oo];
    }
    out[gid] = acc * rv[r];
}

// ---------------------------------------------------------------------------
// Host launcher
// ---------------------------------------------------------------------------
extern "C" void kernel_launch(void* const* d_in, const int* in_sizes, int n_in,
                              void* d_out, int out_size, void* d_ws, size_t ws_size,
                              hipStream_t stream)
{
    const float* x       = (const float*)d_in[0];
    const float* anchors = (const float*)d_in[1];
    const float* w_conv1 = (const float*)d_in[2];
    const float* b_conv1 = (const float*)d_in[3];
    const float* w_loc   = (const float*)d_in[4];
    const float* b_loc   = (const float*)d_in[5];
    const float* w_obj   = (const float*)d_in[6];
    const float* b_obj   = (const float*)d_in[7];
    const float* W1      = (const float*)d_in[8];
    const float* b1      = (const float*)d_in[9];
    const float* W2      = (const float*)d_in[10];
    const float* b2      = (const float*)d_in[11];
    const float* Wc      = (const float*)d_in[12];
    const float* bc      = (const float*)d_in[13];
    const float* Wl      = (const float*)d_in[14];
    const float* bl      = (const float*)d_in[15];
    float* out = (float*)d_out;

    char* p = (char*)d_ws;
    auto alloc = [&](size_t bytes) -> void* {
        void* r = (void*)p;
        p += (bytes + 255) & ~(size_t)255;
        return r;
    };
    float*              wT    = (float*)alloc((size_t)CCH * CCH * 9 * 4);
    float*              xT    = (float*)alloc((size_t)BB * NPIX * CCH * 4);
    float*              h_    = (float*)alloc((size_t)BB * CCH * NPIX * 4);
    float*              boxes = (float*)alloc((size_t)BB * NANC * 4 * 4);
    unsigned long long* keys  = (unsigned long long*)alloc((size_t)BB * NSORT * 8);
    float*              nbx   = (float*)alloc((size_t)BB * PRE_NMS * 4 * 4);
    float*              nar   = (float*)alloc((size_t)BB * PRE_NMS * 4);
    int*                nvd   = (int*)alloc((size_t)BB * PRE_NMS * 4);
    float*              rois  = (float*)alloc((size_t)RTOT * 4 * 4);
    float*              rv    = (float*)alloc((size_t)RTOT * 4);
    float*              feat  = (float*)alloc((size_t)RTOT * FEATK * 4);
    float*              y1    = (float*)alloc((size_t)RTOT * HID * 4);
    float*              y2v   = (float*)alloc((size_t)RTOT * HID * 4);
    (void)ws_size; (void)in_sizes; (void)n_in; (void)out_size;

    // 0) layout transposes (one-time, trivial vs 411MB W1 stream)
    k_wtrans<<<(CCH * CCH * 9 + 255) / 256, 256, 0, stream>>>(w_conv1, wT);
    k_xtrans<<<(BB * CCH * NPIX + 255) / 256, 256, 0, stream>>>(x, xT);

    // 1) conv1 + ReLU (WMMA implicit GEMM)
    k_conv3x3_wmma<<<dim3(CCH / 16, (NPIX + 15) / 16, BB), 32, 0, stream>>>(
        xT, wT, b_conv1, h_);

    // 2) heads + decode + sort keys
    k_heads<<<(BB * NSORT) / 256, 256, 0, stream>>>(
        h_, anchors, w_loc, b_loc, w_obj, b_obj, boxes, keys);

    // 3) bitonic sort (descending) of 2 x 65536 keys
    for (int k = 2; k <= NSORT; k <<= 1)
        for (int j = k >> 1; j > 0; j >>= 1)
            k_bitonic<<<(BB * NSORT) / 256, 256, 0, stream>>>(keys, j, k);

    // 4) greedy NMS, top-300 per batch
    k_nms<<<BB, 1024, 0, stream>>>(keys, boxes, nbx, nar, nvd, rois, rv);

    // 5) ROI-align
    k_roialign<<<((size_t)RTOT * FEATK + 255) / 256, 256, 0, stream>>>(x, rois, feat);

    // 6) FC1 / FC2 (blocked bf16 WMMA, B-tile via TDM)
    const size_t sh = (size_t)(FC_KT * FC_NT + FC_MT * FC_KT) * sizeof(float);
    k_fc_relu_wmma<<<dim3(HID / FC_NT, (RTOT + FC_MT - 1) / FC_MT), 256, sh, stream>>>(
        feat, FEATK, W1, HID, b1, y1, HID, RTOT, HID, FEATK);
    k_fc_relu_wmma<<<dim3(HID / FC_NT, (RTOT + FC_MT - 1) / FC_MT), 256, sh, stream>>>(
        y1, HID, W2, HID, b2, y2v, HID, RTOT, HID, HID);

    // 7) classifier + regressor heads, scale by roi validity
    k_final<<<(RTOT * OUTC + 255) / 256, 256, 0, stream>>>(y2v, Wc, bc, Wl, bl, rv, out);
}
